// Sg2ScVAEModel_74990128988830
// MI455X (gfx1250) — compile-verified, matrix operand order checked by reading, not run on hardware
//
#include <hip/hip_runtime.h>
#include <hip/hip_bf16.h>

// ---------------------------------------------------------------------------
// Types for WMMA (CDNA5 / gfx1250, wave32)
// ---------------------------------------------------------------------------
typedef __attribute__((ext_vector_type(16))) __bf16 v16bf;
typedef __attribute__((ext_vector_type(8)))  __bf16 v8bf;
typedef __attribute__((ext_vector_type(8)))  float  v8f;

#define USE_ASYNC_COPY 1

// ---------------------------------------------------------------------------
// Weight pre-pack: W[K,N] fp32 (row-major)  ->  Wt[N,K] bf16 (row-major)
// ---------------------------------------------------------------------------
__global__ __launch_bounds__(256) void pack_weight(
    const float* __restrict__ W, __bf16* __restrict__ Wt, int K, int N)
{
    __shared__ float tile[32][33];
    const int k0 = blockIdx.x * 32, n0 = blockIdx.y * 32;
    const int t = threadIdx.x;
    {
        const int kk = t >> 3, nn4 = (t & 7) * 4;
        const float4 v = *(const float4*)&W[(size_t)(k0 + kk) * N + n0 + nn4];
        tile[kk][nn4 + 0] = v.x; tile[kk][nn4 + 1] = v.y;
        tile[kk][nn4 + 2] = v.z; tile[kk][nn4 + 3] = v.w;
    }
    __syncthreads();
    {
        const int nn = t >> 3, kk4 = (t & 7) * 4;
        union { __bf16 b[4]; uint2 u; } o;
        o.b[0] = (__bf16)tile[kk4 + 0][nn];
        o.b[1] = (__bf16)tile[kk4 + 1][nn];
        o.b[2] = (__bf16)tile[kk4 + 2][nn];
        o.b[3] = (__bf16)tile[kk4 + 3][nn];
        *(uint2*)&Wt[(size_t)(n0 + nn) * K + k0 + kk4] = o.u;
    }
}

// ---------------------------------------------------------------------------
// Double-buffered bf16-WMMA GEMM:  Y[M,N] = act( X[M,K] @ W[K,N] + bias )
//   - X fp32 row-major, row stride ldx, optional per-row scale (pool mean)
//   - GATHER mode: row r of X is concat(OV[s[r]], PV[r], OV[o[r]]), seg dseg
//   - Wt: pre-packed bf16 [N][K]; B tile DMA'd to LDS with
//     global_load_async_to_lds_b128 (ASYNCcnt) overlapping WMMA compute
//   - fp32 accumulation via v_wmma_f32_16x16x32_bf16
// Tiles: BM=128, BN=128, BK=32; 8 waves in 4(M) x 2(N) grid; wave tile 32x64
// (2 A-frags x 4 B-frags -> 8 WMMAs per K-step); 2-deep LDS pipeline.
// ---------------------------------------------------------------------------
template<bool GATHER>
__global__ __launch_bounds__(256) void gemm_bf16(
    const float* __restrict__ X, int ldx,
    const __bf16* __restrict__ Wt, const float* __restrict__ bias,
    float* __restrict__ Y, int ldy,
    int M, int N, int K, int relu,
    const int* __restrict__ sidx, const int* __restrict__ oidx,
    const float* __restrict__ PV, int ldpv, int dseg,
    const float* __restrict__ rowscale)
{
    constexpr int BM = 128, BN = 128, BK = 32, PK = 40; // PK: padded K stride
    __shared__ __bf16 Abuf[2][BM][PK];   // 2 x 10240 B
    __shared__ __bf16 Bbuf[2][BN][PK];   // 2 x 10240 B (N-major, bf16)

    const int tid  = threadIdx.x;
    const int lane = tid & 31;
    const int wid  = tid >> 5;
    const int wm   = wid >> 1;        // 0..3  (M strip, 32 rows each)
    const int wn   = wid & 1;         // 0..1  (N strip, 64 cols each)
    const int lrow = lane & 15;
    const int half = lane >> 4;

    const int rowBase = blockIdx.y * BM;
    const int colBase = blockIdx.x * BN;

    v8f acc0[4], acc1[4];
    #pragma unroll
    for (int t = 0; t < 4; ++t)
        #pragma unroll
        for (int v = 0; v < 8; ++v) { acc0[t][v] = 0.0f; acc1[t][v] = 0.0f; }

    // A loader: 256 threads cover 128 rows x 32 cols (16 consecutive cols each)
    const int ar  = tid >> 1;          // 0..127
    const int ac  = (tid & 1) * 16;    // 0 or 16
    const int agr = rowBase + ar;
    float ascale = 1.0f;
    const float* abase0 = nullptr;
    int as = 0, ao = 0;
    if (!GATHER) {
        abase0 = X + (size_t)agr * ldx;
        if (rowscale) ascale = rowscale[agr];
    } else {
        as = sidx[agr];
        ao = oidx[agr];
    }

    // B loader: 256 threads cover 128 n-rows x 32 k (16 consecutive k each)
    const int bn2 = tid >> 1;          // 0..127
    const int bkh = (tid & 1) * 16;    // 0 or 16
    const __bf16* wrow = Wt + (size_t)(colBase + bn2) * K + bkh;
#if USE_ASYNC_COPY
    const unsigned bdst0 = (unsigned)(unsigned long long)(const void*)&Bbuf[0][bn2][bkh];
    const unsigned bdst1 = (unsigned)(unsigned long long)(const void*)&Bbuf[1][bn2][bkh];
#endif

    auto stageB = [&](int k0, int buf) {
        const __bf16* wsrc = wrow + k0;
#if USE_ASYNC_COPY
        const unsigned bdst = buf ? bdst1 : bdst0;
        asm volatile(
            "global_load_async_to_lds_b128 %0, %1, off\n\t"
            "global_load_async_to_lds_b128 %0, %1, off offset:16"
            :: "v"(bdst), "v"(wsrc) : "memory");
#else
        uint4 c0 = *(const uint4*)(wsrc);
        uint4 c1 = *(const uint4*)(wsrc + 8);
        *(uint4*)&Bbuf[buf][bn2][bkh]     = c0;
        *(uint4*)&Bbuf[buf][bn2][bkh + 8] = c1;
#endif
    };

    auto loadA = [&](int k0, float4* f) {
        const float* src;
        const int gc = k0 + ac;
        if (!GATHER) {
            src = abase0 + gc;
        } else {
            if (gc < dseg)          src = X  + (size_t)as  * ldx  + gc;
            else if (gc < 2 * dseg) src = PV + (size_t)agr * ldpv + (gc - dseg);
            else                    src = X  + (size_t)ao  * ldx  + (gc - 2 * dseg);
        }
        f[0] = *(const float4*)(src);
        f[1] = *(const float4*)(src + 4);
        f[2] = *(const float4*)(src + 8);
        f[3] = *(const float4*)(src + 12);
    };

    auto storeA = [&](const float4* f, int buf) {
        v8bf lo, hi;
        lo[0] = (__bf16)(f[0].x * ascale); lo[1] = (__bf16)(f[0].y * ascale);
        lo[2] = (__bf16)(f[0].z * ascale); lo[3] = (__bf16)(f[0].w * ascale);
        lo[4] = (__bf16)(f[1].x * ascale); lo[5] = (__bf16)(f[1].y * ascale);
        lo[6] = (__bf16)(f[1].z * ascale); lo[7] = (__bf16)(f[1].w * ascale);
        hi[0] = (__bf16)(f[2].x * ascale); hi[1] = (__bf16)(f[2].y * ascale);
        hi[2] = (__bf16)(f[2].z * ascale); hi[3] = (__bf16)(f[2].w * ascale);
        hi[4] = (__bf16)(f[3].x * ascale); hi[5] = (__bf16)(f[3].y * ascale);
        hi[6] = (__bf16)(f[3].z * ascale); hi[7] = (__bf16)(f[3].w * ascale);
        *(v8bf*)&Abuf[buf][ar][ac]     = lo;
        *(v8bf*)&Abuf[buf][ar][ac + 8] = hi;
    };

    auto compute = [&](int buf) {
        const int arow0 = wm * 32 + lrow;
        const int arow1 = arow0 + 16;
        union { v16bf v; v8bf h[2]; } a0, a1;
        a0.h[0] = *(const v8bf*)&Abuf[buf][arow0][half * 8];
        a0.h[1] = *(const v8bf*)&Abuf[buf][arow0][16 + half * 8];
        a1.h[0] = *(const v8bf*)&Abuf[buf][arow1][half * 8];
        a1.h[1] = *(const v8bf*)&Abuf[buf][arow1][16 + half * 8];
        #pragma unroll
        for (int t = 0; t < 4; ++t) {
            union { v16bf v; v8bf h[2]; } bf_;
            const int n = wn * 64 + t * 16 + lrow;
            bf_.h[0] = *(const v8bf*)&Bbuf[buf][n][half * 8];
            bf_.h[1] = *(const v8bf*)&Bbuf[buf][n][16 + half * 8];
            acc0[t] = __builtin_amdgcn_wmma_f32_16x16x32_bf16(
                false, a0.v, false, bf_.v, (short)0, acc0[t], false, false);
            acc1[t] = __builtin_amdgcn_wmma_f32_16x16x32_bf16(
                false, a1.v, false, bf_.v, (short)0, acc1[t], false, false);
        }
    };

    // ---- pipeline prologue: stage tile 0 into buffer 0 ----
    {
        stageB(0, 0);
        float4 f[4];
        loadA(0, f);
        storeA(f, 0);
#if USE_ASYNC_COPY
        asm volatile("s_wait_asynccnt 0" ::: "memory");
#endif
        __syncthreads();
    }

    const int nt = K / BK;
    for (int i = 0; i < nt; ++i) {
        const int cur = i & 1, nxt = cur ^ 1;
        const bool more = (i + 1) < nt;
        float4 f[4];
        if (more) {
            stageB((i + 1) * BK, nxt);       // async DMA overlaps compute below
            loadA((i + 1) * BK, f);          // global latency overlaps compute
            if (i + 2 < nt) __builtin_prefetch(wrow + (size_t)(i + 2) * BK, 0, 1);
        }
        compute(cur);                        // 8 WMMAs on current buffer
        if (more) storeA(f, nxt);
#if USE_ASYNC_COPY
        asm volatile("s_wait_asynccnt 0" ::: "memory");
#endif
        __syncthreads();
    }

    // ---- epilogue: bias + optional ReLU; C layout VGPR v -> rows v, v+8 ----
    #pragma unroll
    for (int t = 0; t < 4; ++t) {
        const int col = colBase + wn * 64 + t * 16 + lrow;
        const float bv = bias ? bias[col] : 0.0f;
        #pragma unroll
        for (int v = 0; v < 8; ++v) {
            const int row0 = rowBase + wm * 32 + half * 8 + v;
            float y0 = acc0[t][v] + bv;
            float y1 = acc1[t][v] + bv;
            if (relu) { y0 = fmaxf(y0, 0.0f); y1 = fmaxf(y1, 0.0f); }
            Y[(size_t)row0 * ldy + col]        = y0;
            Y[(size_t)(row0 + 16) * ldy + col] = y1;
        }
    }
}

// ---------------------------------------------------------------------------
// Helper kernels
// ---------------------------------------------------------------------------
__global__ void split_triples(const int* __restrict__ t, int* s, int* p, int* o, int n) {
    int i = blockIdx.x * blockDim.x + threadIdx.x;
    if (i < n) { s[i] = t[3 * i]; p[i] = t[3 * i + 1]; o[i] = t[3 * i + 2]; }
}

__global__ void fill_kernel(float* p, int n, float v) {
    int i = blockIdx.x * blockDim.x + threadIdx.x;
    if (i < n) p[i] = v;
}

__global__ void init_ov_box(const int* __restrict__ objs, const float* __restrict__ boxes,
                            const float* __restrict__ emb, const float* __restrict__ W,
                            const float* __restrict__ b, float* __restrict__ ov) {
    int i = blockIdx.x, c = threadIdx.x;
    float v;
    if (c < 128) v = emb[objs[i] * 128 + c];
    else {
        int j = c - 128; float acc = b[j];
        #pragma unroll
        for (int k = 0; k < 6; ++k) acc += boxes[i * 6 + k] * W[k * 128 + j];
        v = acc;
    }
    ov[i * 256 + c] = v;
}

__global__ void init_ov_shape(const int* __restrict__ objs, const float* __restrict__ shapes,
                              const float* __restrict__ emb, const float* __restrict__ W,
                              const float* __restrict__ b, float* __restrict__ ov) {
    int i = blockIdx.x, c = threadIdx.x;
    float v;
    if (c < 128) v = emb[objs[i] * 128 + c];
    else {
        int j = c - 128; float acc = b[j];
        for (int k = 0; k < 128; ++k) acc += shapes[i * 128 + k] * W[k * 128 + j];
        v = acc;
    }
    ov[i * 256 + c] = v;
}

__global__ void init_pv(const int* __restrict__ p, const float* __restrict__ emb,
                        float* __restrict__ pv) {
    int r = blockIdx.x, c = threadIdx.x;               // 256 cols
    pv[r * 256 + c] = emb[p[r] * 256 + c];
}

// pooled[s[r]] += t_out[r, 0:h];  pooled[o[r]] += t_out[r, h+dout : 2h+dout]
__global__ void scatter_pool(const float* __restrict__ tOut, int ldt,
                             const int* __restrict__ s, const int* __restrict__ o,
                             float* __restrict__ pooled, int h, int dout) {
    int r = blockIdx.x;
    int si = s[r], oi = o[r];
    for (int c = threadIdx.x; c < h; c += blockDim.x) {
        atomicAdd(&pooled[(size_t)si * h + c], tOut[(size_t)r * ldt + c]);
        atomicAdd(&pooled[(size_t)oi * h + c], tOut[(size_t)r * ldt + h + dout + c]);
    }
}

__global__ void count_deg(const int* __restrict__ s, const int* __restrict__ o,
                          float* __restrict__ counts, int n) {
    int r = blockIdx.x * blockDim.x + threadIdx.x;
    if (r < n) { atomicAdd(&counts[s[r]], 1.0f); atomicAdd(&counts[o[r]], 1.0f); }
}

__global__ void rscale_kernel(const float* __restrict__ counts, float* __restrict__ rs, int n) {
    int i = blockIdx.x * blockDim.x + threadIdx.x;
    if (i < n) rs[i] = 1.0f / fmaxf(counts[i], 1.0f);
}

__global__ void copy_slice(const float* __restrict__ src, int ls, int off,
                           float* __restrict__ dst, int ld, int w) {
    int r = blockIdx.x;
    for (int c = threadIdx.x; c < w; c += blockDim.x)
        dst[(size_t)r * ld + c] = src[(size_t)r * ls + off + c];
}

// ---------------------------------------------------------------------------
// Host-side orchestration
// ---------------------------------------------------------------------------
struct GconvW {
    const __bf16 *W1, *W2, *W3, *W4;
    const float  *b1, *b2, *b3, *b4;
};

static void launch_gemm(hipStream_t st, const float* X, int ldx, const __bf16* Wt,
                        const float* b, float* Y, int ldy, int M, int N, int K,
                        int relu, const float* rs) {
    dim3 g(N / 128, M / 128);
    gemm_bf16<false><<<g, 256, 0, st>>>(X, ldx, Wt, b, Y, ldy, M, N, K, relu,
                                        nullptr, nullptr, nullptr, 0, 0, rs);
}

// One graph-conv layer. d = d_in = d_out (256 EC, 512 shared), hidden = 512.
static void run_gconv(hipStream_t st, const float* ov_in, int ld_in,
                      const float* pv_in, int ld_pv, int d,
                      float* ov_out, int ld_out, float* tOut,
                      float* h1, float* pooled, float* counts, float* rscale,
                      const int* sA, const int* oA, const GconvW& w) {
    const int T = 4096, O = 1024, Hh = 512;
    const int N2 = 2 * Hh + d;

    { // net1 layer 1: fused gather GEMM  [T, 3d] -> [T, 512], ReLU
        dim3 g(Hh / 128, T / 128);
        gemm_bf16<true><<<g, 256, 0, st>>>(ov_in, ld_in, w.W1, w.b1, h1, Hh,
                                           T, Hh, 3 * d, 1, sA, oA, pv_in, ld_pv, d, nullptr);
    }
    // net1 layer 2: [T, 512] -> [T, 2H+d], ReLU
    launch_gemm(st, h1, Hh, w.W2, w.b2, tOut, N2, T, N2, Hh, 1, nullptr);

    fill_kernel<<<(O * Hh + 255) / 256, 256, 0, st>>>(pooled, O * Hh, 0.0f);
    fill_kernel<<<(O + 255) / 256, 256, 0, st>>>(counts, O, 0.0f);
    scatter_pool<<<T, 256, 0, st>>>(tOut, N2, sA, oA, pooled, Hh, d);
    count_deg<<<(T + 255) / 256, 256, 0, st>>>(sA, oA, counts, T);
    rscale_kernel<<<(O + 255) / 256, 256, 0, st>>>(counts, rscale, O);

    // net2 layer 1: pooled (row-scaled 1/max(count,1)) [O,512] -> [O,512], ReLU
    launch_gemm(st, pooled, Hh, w.W3, w.b3, h1, Hh, O, Hh, Hh, 1, rscale);
    // net2 layer 2: [O,512] -> [O,d], ReLU
    launch_gemm(st, h1, Hh, w.W4, w.b4, ov_out, ld_out, O, d, Hh, 1, nullptr);
}

extern "C" void kernel_launch(void* const* d_in, const int* in_sizes, int n_in,
                              void* d_out, int out_size, void* d_ws, size_t ws_size,
                              hipStream_t stream) {
    (void)in_sizes; (void)n_in; (void)out_size; (void)ws_size;
    const int O = 1024, T = 4096;

    const int*   objs      = (const int*)d_in[0];
    const int*   triples   = (const int*)d_in[1];
    const float* boxes_gt  = (const float*)d_in[2];
    const float* shapes_gt = (const float*)d_in[3];
    const float* emb_box   = (const float*)d_in[5];
    const float* emb_shape = (const float*)d_in[6];
    const float* pemb_box  = (const float*)d_in[7];
    const float* pemb_shape= (const float*)d_in[8];
    const float* box_W     = (const float*)d_in[9];
    const float* box_b     = (const float*)d_in[10];
    const float* shape_W   = (const float*)d_in[11];
    const float* shape_b   = (const float*)d_in[12];

    // ---- workspace layout (all offsets multiple of 256 floats -> 16B aligned)
    float* wsf = (float*)d_ws; size_t off = 0;
    auto A = [&](size_t n) { float* p = wsf + off; off += (n + 255) & ~(size_t)255; return p; };
    int*   sA      = (int*)A(T);
    int*   pA      = (int*)A(T);
    int*   oA      = (int*)A(T);
    float* ov_box  = A((size_t)O * 256);
    float* ov_shape= A((size_t)O * 256);
    float* pv_box  = A((size_t)T * 256);
    float* pv_shape= A((size_t)T * 256);
    float* tA_     = A((size_t)T * 1536);
    float* tB_     = A((size_t)T * 1536);
    float* h1      = A((size_t)T * 512);
    float* pooled  = A((size_t)O * 512);
    float* counts  = A(O);
    float* rscale  = A(O);
    float* ovA     = A((size_t)O * 512);
    float* ovB     = A((size_t)O * 512);
    float* ov_cat  = A((size_t)O * 512);
    float* pv_cat  = A((size_t)T * 512);
    float* hb1     = A((size_t)O * 512);
    float* hb2     = A((size_t)O * 256);
    __bf16* packArena = (__bf16*)A((size_t)10720000);   // ~21.4M bf16 elements

    // ---- pre-pack all GEMM weights to bf16 [N][K] ----
    const __bf16* pw[140] = {};
    size_t poff = 0;
    auto packone = [&](int idx, int K, int N) {
        __bf16* dst = packArena + poff;
        poff += ((size_t)K * N + 255) & ~(size_t)255;
        dim3 g(K / 32, N / 32);
        pack_weight<<<g, 256, 0, stream>>>((const float*)d_in[idx], dst, K, N);
        pw[idx] = dst;
    };
    for (int l = 0; l < 5; ++l) { int b = 13 + l * 8;
        packone(b, 768, 512); packone(b + 2, 512, 1280);
        packone(b + 4, 512, 512); packone(b + 6, 512, 256); }
    for (int l = 0; l < 5; ++l) { int b = 53 + l * 8;
        packone(b, 768, 512); packone(b + 2, 512, 1280);
        packone(b + 4, 512, 512); packone(b + 6, 512, 256); }
    for (int l = 0; l < 3; ++l) { int b = 93 + l * 8;
        packone(b, 1536, 512); packone(b + 2, 512, 1536);
        packone(b + 4, 512, 512); packone(b + 6, 512, 512); }
    packone(117, 256, 512); packone(119, 512, 256);
    packone(121, 256, 128); packone(123, 256, 128);
    packone(125, 256, 512); packone(127, 512, 256);
    packone(129, 256, 128); packone(131, 256, 128);

    auto getw = [&](int b) {
        return GconvW{ pw[b], pw[b + 2], pw[b + 4], pw[b + 6],
                       (const float*)d_in[b + 1], (const float*)d_in[b + 3],
                       (const float*)d_in[b + 5], (const float*)d_in[b + 7] };
    };

    // ---- embeddings / inputs ----
    split_triples<<<(T + 255) / 256, 256, 0, stream>>>(triples, sA, pA, oA, T);
    init_ov_box  <<<O, 256, 0, stream>>>(objs, boxes_gt,  emb_box,   box_W,   box_b,   ov_box);
    init_ov_shape<<<O, 256, 0, stream>>>(objs, shapes_gt, emb_shape, shape_W, shape_b, ov_shape);
    init_pv<<<T, 256, 0, stream>>>(pA, pemb_box,   pv_box);
    init_pv<<<T, 256, 0, stream>>>(pA, pemb_shape, pv_shape);

    // ---- EC box path (5 layers, d=256) ----
    {
        const float* ovi = ov_box;  int ldi = 256;
        const float* pvi = pv_box;  int ldp = 256;
        float* lastT = nullptr;
        for (int l = 0; l < 5; ++l) {
            GconvW w = getw(13 + l * 8);
            float* t = (l & 1) ? tB_ : tA_;
            float* ovo; int ldo;
            if (l == 4) { ovo = ov_cat; ldo = 512; } else { ovo = (l & 1) ? ovB : ovA; ldo = 256; }
            run_gconv(stream, ovi, ldi, pvi, ldp, 256, ovo, ldo, t,
                      h1, pooled, counts, rscale, sA, oA, w);
            ovi = ovo; ldi = ldo; pvi = t + 512; ldp = 1280; lastT = t;
        }
        copy_slice<<<T, 256, 0, stream>>>(lastT, 1280, 512, pv_cat, 512, 256);
    }

    // ---- EC shape path (5 layers, d=256) ----
    {
        const float* ovi = ov_shape; int ldi = 256;
        const float* pvi = pv_shape; int ldp = 256;
        float* lastT = nullptr;
        for (int l = 0; l < 5; ++l) {
            GconvW w = getw(53 + l * 8);
            float* t = (l & 1) ? tB_ : tA_;
            float* ovo; int ldo;
            if (l == 4) { ovo = ov_cat + 256; ldo = 512; } else { ovo = (l & 1) ? ovB : ovA; ldo = 256; }
            run_gconv(stream, ovi, ldi, pvi, ldp, 256, ovo, ldo, t,
                      h1, pooled, counts, rscale, sA, oA, w);
            ovi = ovo; ldi = ldo; pvi = t + 512; ldp = 1280; lastT = t;
        }
        copy_slice<<<T, 256, 0, stream>>>(lastT, 1280, 512, pv_cat + 256, 512, 256);
    }

    // ---- shared path (3 layers, d=512) ----
    float* ov_fin = nullptr;
    {
        const float* ovi = ov_cat; int ldi = 512;
        const float* pvi = pv_cat; int ldp = 512;
        for (int l = 0; l < 3; ++l) {
            GconvW w = getw(93 + l * 8);
            float* t = (l & 1) ? tB_ : tA_;
            float* ovo = (l == 1) ? ovB : ovA;
            run_gconv(stream, ovi, ldi, pvi, ldp, 512, ovo, 512, t,
                      h1, pooled, counts, rscale, sA, oA, w);
            ovi = ovo; ldi = 512; pvi = t + 512; ldp = 1536;
        }
        ov_fin = ovA;
    }

    // ---- heads: output columns [mu_box | mu_shape | lv_box | lv_shape] ----
    float* out = (float*)d_out;
    // box: ov_fin[:, :256]
    launch_gemm(stream, ov_fin,       512, pw[117], (const float*)d_in[118], hb1, 512, O, 512, 256, 1, nullptr);
    launch_gemm(stream, hb1,          512, pw[119], (const float*)d_in[120], hb2, 256, O, 256, 512, 1, nullptr);
    launch_gemm(stream, hb2,          256, pw[121], (const float*)d_in[122], out + 0,   512, O, 128, 256, 0, nullptr);
    launch_gemm(stream, hb2,          256, pw[123], (const float*)d_in[124], out + 256, 512, O, 128, 256, 0, nullptr);
    // shape: ov_fin[:, 256:512]
    launch_gemm(stream, ov_fin + 256, 512, pw[125], (const float*)d_in[126], hb1, 512, O, 512, 256, 1, nullptr);
    launch_gemm(stream, hb1,          512, pw[127], (const float*)d_in[128], hb2, 256, O, 256, 512, 1, nullptr);
    launch_gemm(stream, hb2,          256, pw[129], (const float*)d_in[130], out + 128, 512, O, 128, 256, 0, nullptr);
    launch_gemm(stream, hb2,          256, pw[131], (const float*)d_in[132], out + 384, 512, O, 128, 256, 0, nullptr);
}
